// RNNModel_46540265619727
// MI455X (gfx1250) — matrix operand order repaired and checked by reading, not compiled, not run on hardware
//
#include <hip/hip_runtime.h>
#include <hip/hip_bf16.h>

typedef __attribute__((ext_vector_type(16))) _Float16 v16h;
typedef __attribute__((ext_vector_type(8)))  _Float16 v8h;
typedef __attribute__((ext_vector_type(8)))  float    v8f;
typedef __attribute__((ext_vector_type(16))) float    v16f;

namespace {
constexpr int kB = 64, kT = 2048, kI = 64, kH = 256;
constexpr int kMT = 16;                // batch tile = WMMA M
constexpr int kWaves = 16;             // 1 N-tile (16 channels) per wave
constexpr int kThreads = kWaves * 32;  // 512 threads, wave32
}

__device__ __forceinline__ float fast_tanh(float v) {
#if __has_builtin(__builtin_amdgcn_tanhf)
  return __builtin_amdgcn_tanhf(v);      // v_tanh_f32 (TRANS32, branch-free)
#else
  // branch-free fallback: tanh(v) = 1 - 2/(e^{2v}+1)
  const float e = __expf(v + v);
  return 1.f - 2.f * __builtin_amdgcn_rcpf(e + 1.f);
#endif
}

// 16 consecutive f32 -> v16h (B fragment: row of W, contiguous along K)
__device__ __forceinline__ v16h cvt16(const float* __restrict__ p) {
  const v16f v = *(const v16f*)p;
  v16h r;
#pragma unroll
  for (int e = 0; e < 16; ++e) r[e] = (_Float16)v[e];
  return r;
}

// A fragment halves: elements 0..7 at `lo`, 8..15 at `hi` (K split per ISA layout)
__device__ __forceinline__ v16h cvt8x2(const float* __restrict__ lo,
                                       const float* __restrict__ hi) {
  const v8f a = *(const v8f*)lo;
  const v8f b = *(const v8f*)hi;
  v16h r;
#pragma unroll
  for (int e = 0; e < 8; ++e) { r[e] = (_Float16)a[e]; r[e + 8] = (_Float16)b[e]; }
  return r;
}

__global__ __launch_bounds__(kThreads) void rnn_fused(
    const float* __restrict__ x,   const float* __restrict__ Wih,
    const float* __restrict__ Whh, const float* __restrict__ bih,
    const float* __restrict__ bhh, const float* __restrict__ Wfc,
    const float* __restrict__ bfc, float* __restrict__ out) {
  __shared__ __align__(16) _Float16 hbuf[2][kMT * kH];  // double-buffered h, f16

  const int tid   = threadIdx.x;
  const int wid   = tid >> 5;    // 0..15: N-tile owned by this wave
  const int lane  = tid & 31;
  const int lhalf = lane >> 4;   // 0/1: lane half selects K sub-range (ISA A/B layout)
  const int l16   = lane & 15;
  const int bt    = blockIdx.x;  // batch tile (4 tiles of 16)

  // ---- W_hh / W_ih B fragments resident in VGPRs (f16), loaded once ----
  // out[m,n] = sum_k h[m,k] * W[n,k]  =>  B[k][n] = W[n][k] (contiguous along k)
  const int n = wid * 16 + l16;  // this lane's output channel
  v16h bh[8];                    // 64 VGPRs
#pragma unroll
  for (int kc = 0; kc < 8; ++kc)
    bh[kc] = cvt8x2(Whh + n * kH + kc * 32 + lhalf * 8, Whh + n * kH + kc * 32 + lhalf * 8 + 16);
  v16h bi[2];                    // 16 VGPRs
#pragma unroll
  for (int kc = 0; kc < 2; ++kc)
    bi[kc] = cvt8x2(Wih + n * kI + kc * 32 + lhalf * 8, Wih + n * kI + kc * 32 + lhalf * 8 + 16);
  const float bias = bih[n] + bhh[n];

  // Output head: wave wid owns row m = wid; lane covers channels lane*8..lane*8+7
  float wf[8];
#pragma unroll
  for (int e = 0; e < 8; ++e) wf[e] = Wfc[lane * 8 + e];
  const float bias_fc = bfc[0];

  // h(0) = 0
#pragma unroll
  for (int q = 0; q < (kMT * kH) / kThreads; ++q)
    hbuf[0][tid + q * kThreads] = (_Float16)0.f;

  const float* xrow = x + (size_t)(bt * kMT + l16) * kT * kI;  // A row: M = lane&15

  // preload x_0 fragments (software pipeline)
  v16h ax[2];
#pragma unroll
  for (int kc = 0; kc < 2; ++kc) {
    const float* px = xrow + kc * 32 + lhalf * 8;
    ax[kc] = cvt8x2(px, px + 16);
  }
  __syncthreads();

  for (int t = 0; t < kT; ++t) {
    const int p = t & 1;
    const _Float16* __restrict__ hrd = &hbuf[p][l16 * kH + lhalf * 8];

    // A fragment loader for K-chunk kc of h (two 16B LDS reads per lane)
    auto loadA = [&](int kc) -> v16h {
      const _Float16* ph = hrd + kc * 32;
      const v8h lo = *(const v8h*)ph;
      const v8h hi = *(const v8h*)(ph + 16);
      return __builtin_shufflevector(lo, hi, 0, 1, 2, 3, 4, 5, 6, 7,
                                     8, 9, 10, 11, 12, 13, 14, 15);
    };

    // dual accumulators: break the serial WMMA->WMMA C dependence
    v8f c0, c1;
#pragma unroll
    for (int r = 0; r < 8; ++r) { c0[r] = bias; c1[r] = 0.f; }

    // pre = x_t @ W_ih^T  (K = 64 -> 2 chunks, A already in registers)
    c0 = __builtin_amdgcn_wmma_f32_16x16x32_f16(false, ax[0], false, bi[0],
                                                (short)0, c0, false, false);
    c1 = __builtin_amdgcn_wmma_f32_16x16x32_f16(false, ax[1], false, bi[1],
                                                (short)0, c1, false, false);

    // + h @ W_hh^T  (K = 256 -> 8 chunks), 2-deep pipelined LDS A fragments
    v16h ahp[3];
    ahp[0] = loadA(0);
    ahp[1] = loadA(1);
#pragma unroll
    for (int kc = 0; kc < 8; ++kc) {
      if (kc + 2 < 8) ahp[(kc + 2) % 3] = loadA(kc + 2);
      if (kc & 1)
        c1 = __builtin_amdgcn_wmma_f32_16x16x32_f16(false, ahp[kc % 3], false,
                                                    bh[kc], (short)0, c1, false, false);
      else
        c0 = __builtin_amdgcn_wmma_f32_16x16x32_f16(false, ahp[kc % 3], false,
                                                    bh[kc], (short)0, c0, false, false);
    }

    // tanh (native v_tanh_f32) + stage h(t+1) into the other LDS buffer
    // C layout: VGPR r, lanes 0-15 -> M=r, lanes 16-31 -> M=8+r; N = wid*16+(lane&15)
    _Float16* hw = hbuf[1 - p];
#pragma unroll
    for (int r = 0; r < 8; ++r)
      hw[(lhalf * 8 + r) * kH + n] = (_Float16)fast_tanh(c0[r] + c1[r]);

    // branchless prefetch of next step's x fragments (overlaps the barrier)
    {
      const int tn = (t + 1 < kT) ? (t + 1) : t;
#pragma unroll
      for (int kc = 0; kc < 2; ++kc) {
        const float* px = xrow + (size_t)tn * kI + kc * 32 + lhalf * 8;
        ax[kc] = cvt8x2(px, px + 16);
      }
    }
    __syncthreads();

    // output head: out[b, t] = dot(h(t+1)[wid, :], W_fc) + b_fc  (full-wave reduce)
    {
      const _Float16* ph = &hbuf[1 - p][wid * kH + lane * 8];
      const v8h hv = *(const v8h*)ph;
      float s = 0.f;
#pragma unroll
      for (int e = 0; e < 8; ++e) s += (float)hv[e] * wf[e];
#pragma unroll
      for (int off = 16; off >= 1; off >>= 1) s += __shfl_xor(s, off, 32);
      if (lane == 0) out[(size_t)(bt * kMT + wid) * kT + t] = s + bias_fc;
    }
    // single barrier per step: next iteration writes hbuf[p], whose last readers
    // (A fragments of step t) are fenced by this barrier; out-head reads hbuf[1-p],
    // which is not written until step t+2, after the t+1 barrier.
  }
}

extern "C" void kernel_launch(void* const* d_in, const int* in_sizes, int n_in,
                              void* d_out, int out_size, void* d_ws, size_t ws_size,
                              hipStream_t stream) {
  (void)in_sizes; (void)n_in; (void)d_ws; (void)ws_size; (void)out_size;
  const float* x   = (const float*)d_in[0];
  const float* Wih = (const float*)d_in[1];
  const float* Whh = (const float*)d_in[2];
  const float* bih = (const float*)d_in[3];
  const float* bhh = (const float*)d_in[4];
  const float* Wfc = (const float*)d_in[5];
  const float* bfc = (const float*)d_in[6];
  rnn_fused<<<kB / kMT, kThreads, 0, stream>>>(x, Wih, Whh, bih, bhh, Wfc, bfc,
                                               (float*)d_out);
}
